// LocalGlobalAttention_1975684956479
// MI455X (gfx1250) — matrix operand order repaired and checked by reading, not compiled, not run on hardware
//
#include <hip/hip_runtime.h>
#include <hip/hip_bf16.h>
#include <math.h>

// ---------------------------------------------------------------------------
// Problem constants (from reference)
// ---------------------------------------------------------------------------
#define BB     2
#define NN     2048
#define DD     1024
#define HH     16
#define HD     64
#define WIN    128
#define GSTRIDE 64
#define MM     256
#define RANK   64
#define LORA   4
#define ROWS   (BB * NN)          // 4096
#define SCALE  0.125f             // 64^-0.5

typedef __attribute__((ext_vector_type(16))) _Float16 v16h;
typedef __attribute__((ext_vector_type(8)))  float    v8f;

union Frag16 {
    v16h     v;
    _Float16 h[16];
    uint4    q[2];
};

__device__ __forceinline__ v8f wmma_f16(const Frag16& a, const Frag16& b, v8f c) {
    return __builtin_amdgcn_wmma_f32_16x16x32_f16(
        /*neg_a=*/false, a.v, /*neg_b=*/false, b.v,
        /*c_mod=*/(short)0, c, /*reuse_a=*/false, /*reuse_b=*/false);
}

// ---------------------------------------------------------------------------
// Swizzled B-matrix storage for v_wmma_f32_16x16x32_f16.
// Tile (jb: 16 cols, kb: 32 K-rows). Lane = jn + 16*(kr>=16), lane holds 16
// contiguous f16 (K = kb*32 + (kr>=16?16:0) + 0..15). One lane fragment = 32B.
// f16 index = ((jb*nKb + kb)*32 + lane)*16 + (kr & 15)
// ---------------------------------------------------------------------------
__device__ __forceinline__ size_t wsw_index(int k, int j, int nKb) {
    int kb = k >> 5, kr = k & 31;
    int jb = j >> 4, jn = j & 15;
    int lane = jn + ((kr >= 16) ? 16 : 0);
    return (((size_t)(jb * nKb + kb) * 32) + lane) * 16 + (size_t)(kr & 15);
}

// ---------------------------------------------------------------------------
// K1: W_eff = (U * alpha) @ V^T + A @ Bm   (D x D), stored swizzled f16
// ---------------------------------------------------------------------------
__global__ void lga_build_weff(const float* __restrict__ U, const float* __restrict__ V,
                               const float* __restrict__ alpha, const float* __restrict__ A,
                               const float* __restrict__ Bm, _Float16* __restrict__ Wsw) {
    int idx = blockIdx.x * blockDim.x + threadIdx.x;
    if (idx >= DD * DD) return;
    int i = idx >> 10;          // K index (input dim)
    int j = idx & (DD - 1);     // N index (output dim)
    float s = 0.f;
#pragma unroll 8
    for (int r = 0; r < RANK; ++r)
        s += U[i * RANK + r] * alpha[r] * V[j * RANK + r];
#pragma unroll
    for (int l = 0; l < LORA; ++l)
        s += A[i * LORA + l] * Bm[l * DD + j];
    Wsw[wsw_index(i, j, DD / 32)] = (_Float16)s;
}

// K1b: out_W (D x D, row = out, col = in) -> swizzled f16 with B[i][j] = W[j][i]
__global__ void lga_build_wout(const float* __restrict__ W, _Float16* __restrict__ Wsw) {
    int idx = blockIdx.x * blockDim.x + threadIdx.x;
    if (idx >= DD * DD) return;
    int i = idx >> 10;
    int j = idx & (DD - 1);
    Wsw[wsw_index(i, j, DD / 32)] = (_Float16)W[j * DD + i];
}

// K2: f32 -> f16 conversion
__global__ void lga_f32_to_f16(const float* __restrict__ src, _Float16* __restrict__ dst, int n) {
    int i = blockIdx.x * blockDim.x + threadIdx.x;
    if (i < n) dst[i] = (_Float16)src[i];
}

// ---------------------------------------------------------------------------
// K3 / K5: GEMM  C(ROWS x DD) = X(ROWS x DD) @ Wsw + bias
// One wave per 16x64 output strip: a single A fragment is reused across 4
// B tiles (4 WMMAs per K-step), cutting A-fragment L2 traffic 4x.
// MODE 0: write f16 into (B,H,N,HD) layout (for attention consumption)
// MODE 1: write f32 row-major into d_out
// ---------------------------------------------------------------------------
template <int MODE>
__global__ __launch_bounds__(32)
void lga_gemm_wmma(const _Float16* __restrict__ X, const _Float16* __restrict__ Wsw,
                   const float* __restrict__ bias, void* __restrict__ out) {
    const int lane = threadIdx.x;
    const int jb4 = blockIdx.x;   // 0..15  : column strip (4 tiles = 64 cols)
    const int rb  = blockIdx.y;   // 0..255 : row tile
    const int nKb = DD / 32;      // 32

    const int arow  = rb * 16 + (lane & 15);
    const int koffA = ((lane >> 4) & 1) * 16;
    const _Float16* xrow = X + (size_t)arow * DD + koffA;
    // fragment base pointers for the 4 column tiles
    const _Float16* wp[4];
#pragma unroll
    for (int c = 0; c < 4; ++c)
        wp[c] = Wsw + ((size_t)((jb4 * 4 + c) * nKb) * 32 + lane) * 16;

    v8f acc[4] = {{}, {}, {}, {}};
#pragma unroll 2
    for (int kb = 0; kb < nKb; ++kb) {
        Frag16 a;
        const _Float16* ax = xrow + kb * 32;
        a.q[0] = *(const uint4*)(ax);
        a.q[1] = *(const uint4*)(ax + 8);
        __builtin_prefetch(wp[0] + 32 * 16, 0, 1);   // next K-tile -> global_prefetch
#pragma unroll
        for (int c = 0; c < 4; ++c) {
            Frag16 b;
            b.q[0] = *(const uint4*)(wp[c]);
            b.q[1] = *(const uint4*)(wp[c] + 8);
            wp[c] += 32 * 16;
            acc[c] = wmma_f16(a, b, acc[c]);
        }
    }

    const int rhalf = (lane < 16) ? 0 : 8;
#pragma unroll
    for (int c = 0; c < 4; ++c) {
        const int col = jb4 * 64 + c * 16 + (lane & 15);
        const float bj = bias[col];
#pragma unroll
        for (int r = 0; r < 8; ++r) {
            int gr = rb * 16 + rhalf + r;        // global row = b*NN + n
            float val = acc[c][r] + bj;
            if (MODE == 0) {
                int bI = gr >> 11, n = gr & (NN - 1);
                int h = col >> 6, hd = col & 63;
                ((_Float16*)out)[(((size_t)(bI * HH + h) * NN + n) * HD) + hd] = (_Float16)val;
            } else {
                ((float*)out)[(size_t)gr * DD + col] = val;
            }
        }
    }
}

// ---------------------------------------------------------------------------
// K4: attention. One wave per (b*H + h, 16-query tile).
// Online-softmax over key tiles; QK^T and PV via WMMA.
// ---------------------------------------------------------------------------
__device__ __forceinline__ void lga_attn_phase(
    const Frag16 qa[2], const _Float16* __restrict__ kbase,
    const _Float16* __restrict__ vbase, int nkeys, int keystride, int kstart,
    bool masked, int q0, v8f comb[4], int lane,
    float (&Pt)[16][17], _Float16 (&vT)[64][16]) {

    v8f O[4] = {{}, {}, {}, {}};
    float m[8], l[8];
#pragma unroll
    for (int r = 0; r < 8; ++r) { m[r] = -1e30f; l[r] = 0.f; }

    const int ntiles = (nkeys + 15) >> 4;
    for (int t = 0; t < ntiles; ++t) {
        const int kt0 = t * 16;
        // ---- scores S = q . k over HD=64 (2 WMMAs) ----
        v8f S = {};
        {
            int kc = lane & 15;
            int key = kt0 + kc;
            bool valid = key < nkeys;
            int keyg = kstart + (valid ? key : 0) * keystride;
            const _Float16* krow = kbase + (size_t)keyg * HD + (((lane >> 4) & 1) * 16);
#pragma unroll
            for (int kb = 0; kb < 2; ++kb) {
                Frag16 bf;
                if (valid) {
                    bf.q[0] = *(const uint4*)(krow + kb * 32);
                    bf.q[1] = *(const uint4*)(krow + kb * 32 + 8);
                } else {
                    bf.q[0] = make_uint4(0, 0, 0, 0);
                    bf.q[1] = make_uint4(0, 0, 0, 0);
                }
                S = wmma_f16(qa[kb], bf, S);
            }
        }
        // ---- mask + online softmax (row stats via half-wave shuffles) ----
        const int col = lane & 15;
        const int keygcol = kstart + (kt0 + col) * keystride;
        float p[8];
#pragma unroll
        for (int r = 0; r < 8; ++r) {
            int rowg = q0 + ((lane < 16) ? r : r + 8);
            float x = S[r] * SCALE;
            if (masked) {
                int d = rowg - keygcol; if (d < 0) d = -d;
                if (d > WIN) x = -1e30f;
            }
            if (kt0 + col >= nkeys) x = -1e30f;
            float rm = x;
#pragma unroll
            for (int off = 1; off < 16; off <<= 1)
                rm = fmaxf(rm, __shfl_xor(rm, off, 32));
            float mnew = fmaxf(m[r], rm);
            float corr = __expf(m[r] - mnew);
            m[r] = mnew;
            float pe = __expf(x - mnew);
            float rs = pe;
#pragma unroll
            for (int off = 1; off < 16; off <<= 1)
                rs += __shfl_xor(rs, off, 32);
            l[r] = l[r] * corr + rs;
#pragma unroll
            for (int c = 0; c < 4; ++c) O[c][r] *= corr;
            p[r] = pe;
        }
        // ---- P -> LDS (for A-fragment transpose) ----
#pragma unroll
        for (int r = 0; r < 8; ++r) {
            int rowm = (lane < 16) ? r : r + 8;
            Pt[rowm][col] = p[r];
        }
        // ---- V tile transposed into LDS: vT[hd][key] ----
        for (int i = lane; i < 16 * HD; i += 32) {
            int key = i >> 6, hd = i & 63;
            _Float16 val = (_Float16)0.f;
            if (kt0 + key < nkeys) {
                int keyg = kstart + (kt0 + key) * keystride;
                val = vbase[(size_t)keyg * HD + hd];
            }
            vT[hd][key] = val;
        }
        __builtin_amdgcn_wave_barrier();
        // ---- A fragment from P (K = 16 keys, upper 16 K zero) ----
        Frag16 pa;
        {
            int rowm = lane & 15;
#pragma unroll
            for (int i = 0; i < 16; ++i) pa.h[i] = (lane < 16) ? (_Float16)Pt[rowm][i] : (_Float16)0.f;
        }
        // ---- O += P @ V, 4 column chunks of HD ----
#pragma unroll
        for (int c = 0; c < 4; ++c) {
            Frag16 bv;
            if (lane < 16) {
                const _Float16* src = &vT[c * 16 + lane][0];
                bv.q[0] = *(const uint4*)(src);
                bv.q[1] = *(const uint4*)(src + 8);
            } else {  // K=16..31 rows of B are zero (only 16 keys)
                bv.q[0] = make_uint4(0, 0, 0, 0);
                bv.q[1] = make_uint4(0, 0, 0, 0);
            }
            O[c] = wmma_f16(pa, bv, O[c]);
        }
        __builtin_amdgcn_wave_barrier();
    }
    // ---- normalize and accumulate into combined ----
#pragma unroll
    for (int r = 0; r < 8; ++r) {
        float inv = (l[r] > 0.f) ? 1.f / l[r] : 0.f;
#pragma unroll
        for (int c = 0; c < 4; ++c) comb[c][r] += O[c][r] * inv;
    }
}

__global__ __launch_bounds__(32)
void lga_attn(const _Float16* __restrict__ qh, const _Float16* __restrict__ kh,
              const _Float16* __restrict__ vh, const _Float16* __restrict__ memh,
              _Float16* __restrict__ fused) {
    __shared__ float Pt[16][17];
    __shared__ __align__(16) _Float16 vT[64][16];

    const int lane = threadIdx.x;
    const int bh = blockIdx.y;              // b*HH + h
    const int b = bh >> 4, h = bh & 15;
    const int q0 = blockIdx.x * 16;

    const _Float16* qp = qh + (size_t)bh * NN * HD;
    const _Float16* kp = kh + (size_t)bh * NN * HD;
    const _Float16* vp = vh + (size_t)bh * NN * HD;
    const _Float16* mp = memh + (size_t)b * MM * HD;

    // Q A-fragments over K = 0..63 (two 32-wide chunks)
    Frag16 qa[2];
    {
        int row = q0 + (lane & 15);
        int koff = ((lane >> 4) & 1) * 16;
        const _Float16* qr = qp + (size_t)row * HD + koff;
#pragma unroll
        for (int kb = 0; kb < 2; ++kb) {
            qa[kb].q[0] = *(const uint4*)(qr + kb * 32);
            qa[kb].q[1] = *(const uint4*)(qr + kb * 32 + 8);
        }
    }

    v8f comb[4] = {{}, {}, {}, {}};

    // local windowed attention (|i-j| <= WIN), keys [klo, khi)
    int klo = q0 - WIN; if (klo < 0) klo = 0;
    int khi = q0 + 16 + WIN; if (khi > NN) khi = NN;
    lga_attn_phase(qa, kp, vp, khi - klo, 1, klo, true, q0, comb, lane, Pt, vT);
    // strided global attention: 32 keys at n = g*GSTRIDE
    lga_attn_phase(qa, kp, vp, NN / GSTRIDE, GSTRIDE, 0, false, q0, comb, lane, Pt, vT);
    // retrieval over memory vectors (K and V both = memory)
    lga_attn_phase(qa, mp, mp, MM, 1, 0, false, q0, comb, lane, Pt, vT);

    // store fused[(b*NN+n)][h*64 + hd] as f16
#pragma unroll
    for (int c = 0; c < 4; ++c) {
#pragma unroll
        for (int r = 0; r < 8; ++r) {
            int rowg = q0 + ((lane < 16) ? r : r + 8);
            int hd = c * 16 + (lane & 15);
            fused[((size_t)(b * NN + rowg)) * DD + h * HD + hd] = (_Float16)comb[c][r];
        }
    }
}

// ---------------------------------------------------------------------------
// K6: small heads from head 0 of combined (= fused[:, 0:64])
// outputs concatenated: out(4194304) | temporal(8192) | identity(262144) | citation(131072)
// ---------------------------------------------------------------------------
__global__ void lga_heads(const _Float16* __restrict__ fused,
                          const float* __restrict__ tW, const float* __restrict__ tb,
                          const float* __restrict__ iW, const float* __restrict__ ib,
                          const float* __restrict__ cW, const float* __restrict__ cb,
                          float* __restrict__ out) {
    int gid = blockIdx.x * blockDim.x + threadIdx.x;
    if (gid >= ROWS * 98) return;
    int row = gid / 98;
    int o = gid - row * 98;
    const _Float16* f = fused + (size_t)row * DD;  // head 0 = first 64 cols
    const float* w; float bias; float* dst;
    const size_t OFF_T = (size_t)ROWS * DD;                 // 4194304
    const size_t OFF_I = OFF_T + (size_t)ROWS * 2;          // 4202496
    const size_t OFF_C = OFF_I + (size_t)ROWS * 64;         // 4464640
    if (o < 2)       { w = tW + o * HD;        bias = tb[o];      dst = out + OFF_T + (size_t)row * 2  + o; }
    else if (o < 66) { int k = o - 2;  w = iW + k * HD; bias = ib[k]; dst = out + OFF_I + (size_t)row * 64 + k; }
    else             { int k = o - 66; w = cW + k * HD; bias = cb[k]; dst = out + OFF_C + (size_t)row * 32 + k; }
    float s = bias;
#pragma unroll 8
    for (int d = 0; d < HD; ++d) s += (float)f[d] * w[d];
    *dst = s;
}

// ---------------------------------------------------------------------------
// Launcher
// ---------------------------------------------------------------------------
extern "C" void kernel_launch(void* const* d_in, const int* in_sizes, int n_in,
                              void* d_out, int out_size, void* d_ws, size_t ws_size,
                              hipStream_t stream) {
    (void)in_sizes; (void)n_in; (void)out_size; (void)ws_size;
    const float* x     = (const float*)d_in[0];
    const float* mem   = (const float*)d_in[1];
    const float* qU = (const float*)d_in[2],  *qV = (const float*)d_in[3],
               * qa = (const float*)d_in[4],  *qA = (const float*)d_in[5],
               * qB = (const float*)d_in[6],  *qb = (const float*)d_in[7];
    const float* kU = (const float*)d_in[8],  *kV = (const float*)d_in[9],
               * ka = (const float*)d_in[10], *kA = (const float*)d_in[11],
               * kB = (const float*)d_in[12], *kb = (const float*)d_in[13];
    const float* vU = (const float*)d_in[14], *vV = (const float*)d_in[15],
               * va = (const float*)d_in[16], *vA = (const float*)d_in[17],
               * vB = (const float*)d_in[18], *vb = (const float*)d_in[19];
    const float* outW = (const float*)d_in[20], *outb = (const float*)d_in[21];
    const float* tW = (const float*)d_in[22], *tb = (const float*)d_in[23];
    const float* iW = (const float*)d_in[24], *ib = (const float*)d_in[25];
    const float* cW = (const float*)d_in[26], *cb = (const float*)d_in[27];
    float* out = (float*)d_out;

    // workspace carve-up (all offsets 256B aligned)
    char* base = (char*)d_ws;
    size_t off = 0;
    auto carve = [&](size_t bytes) {
        char* p = base + off;
        off = (off + bytes + 255) & ~(size_t)255;
        return p;
    };
    const size_t WBYTES = (size_t)DD * DD * sizeof(_Float16);   // 2 MB
    _Float16* Wq = (_Float16*)carve(WBYTES);
    _Float16* Wk = (_Float16*)carve(WBYTES);
    _Float16* Wv = (_Float16*)carve(WBYTES);
    _Float16* Wo = (_Float16*)carve(WBYTES);
    _Float16* xh   = (_Float16*)carve((size_t)ROWS * DD * sizeof(_Float16));
    _Float16* qh   = (_Float16*)carve((size_t)ROWS * DD * sizeof(_Float16));
    _Float16* khb  = (_Float16*)carve((size_t)ROWS * DD * sizeof(_Float16));
    _Float16* vhb  = (_Float16*)carve((size_t)ROWS * DD * sizeof(_Float16));
    _Float16* fus  = (_Float16*)carve((size_t)ROWS * DD * sizeof(_Float16));
    _Float16* memh = (_Float16*)carve((size_t)BB * MM * HD * sizeof(_Float16));

    // 1) effective weights (swizzled f16)
    dim3 wgrid((DD * DD + 255) / 256), wblk(256);
    lga_build_weff<<<wgrid, wblk, 0, stream>>>(qU, qV, qa, qA, qB, Wq);
    lga_build_weff<<<wgrid, wblk, 0, stream>>>(kU, kV, ka, kA, kB, Wk);
    lga_build_weff<<<wgrid, wblk, 0, stream>>>(vU, vV, va, vA, vB, Wv);
    lga_build_wout<<<wgrid, wblk, 0, stream>>>(outW, Wo);

    // 2) f16 conversions
    lga_f32_to_f16<<<(ROWS * DD + 255) / 256, 256, 0, stream>>>(x, xh, ROWS * DD);
    lga_f32_to_f16<<<(BB * MM * HD + 255) / 256, 256, 0, stream>>>(mem, memh, BB * MM * HD);

    // 3) q/k/v projections (WMMA GEMMs) -> (B,H,N,HD) f16
    dim3 ggrid(DD / 64, ROWS / 16), gblk(32);
    lga_gemm_wmma<0><<<ggrid, gblk, 0, stream>>>(xh, Wq, qb, (void*)qh);
    lga_gemm_wmma<0><<<ggrid, gblk, 0, stream>>>(xh, Wk, kb, (void*)khb);
    lga_gemm_wmma<0><<<ggrid, gblk, 0, stream>>>(xh, Wv, vb, (void*)vhb);

    // 4) attention (local + strided-global + retrieval), fused f16
    dim3 agrid(NN / 16, BB * HH), ablk(32);
    lga_attn<<<agrid, ablk, 0, stream>>>(qh, khb, vhb, memh, fus);

    // 5) output projection -> d_out[0 .. ROWS*DD) f32
    lga_gemm_wmma<1><<<ggrid, gblk, 0, stream>>>(fus, Wo, outb, (void*)out);

    // 6) temporal / identity / citation heads
    lga_heads<<<(ROWS * 98 + 255) / 256, 256, 0, stream>>>(fus, tW, tb, iW, ib, cW, cb, out);
}
